// MptAttention_13451837571602
// MI455X (gfx1250) — matrix operand order repaired and checked by reading, not compiled, not run on hardware
//
#include <hip/hip_runtime.h>
#include <hip/hip_bf16.h>

// ---- problem constants (from reference) ----
constexpr int kB  = 2;
constexpr int kS  = 2048;
constexpr int kH  = 2048;
constexpr int kNH = 16;
constexpr int kHD = 128;
constexpr float kScale = 0.08838834764831845f;   // 1/sqrt(128)
constexpr float kClip  = 8.0f;

typedef __attribute__((ext_vector_type(16))) _Float16 v16h;
typedef __attribute__((ext_vector_type(8)))  _Float16 v8h;
typedef __attribute__((ext_vector_type(8)))  float    v8f;
typedef __attribute__((ext_vector_type(4)))  float    v4f;

union AFrag { v16h v; v8h h[2]; };

__device__ __forceinline__ v8f wmma16(const AFrag a, const AFrag b, v8f c) {
  // D = A(16x32 f16) x B(32x16 f16) + C(16x16 f32)
  return __builtin_amdgcn_wmma_f32_16x16x32_f16(
      /*neg_a=*/false, a.v, /*neg_b=*/false, b.v,
      /*c_mod=*/(short)0, c, /*reuse_a=*/false, /*reuse_b=*/false);
}

// ---- CDNA5 async global->LDS DMA (ASYNCcnt-tracked) ----
// LDS address operand: low 32 bits of a flat shared pointer are the LDS
// byte offset (ISA 10.2).
__device__ __forceinline__ void async_ld_b128(void* lds, const void* gsrc) {
  asm volatile("global_load_async_to_lds_b128 %0, %1, off"
               :: "v"((unsigned)(unsigned long long)(uintptr_t)lds),
                  "v"((unsigned long long)(uintptr_t)gsrc)
               : "memory");
}
__device__ __forceinline__ void wait_async0() {
  asm volatile("s_wait_asynccnt 0x0" ::: "memory");
}

// ---- staged global->LDS helpers (fp32 source, f16 LDS) ----
__device__ __forceinline__ void gload16(const float* __restrict__ src, v4f* p) {
#pragma unroll
  for (int i = 0; i < 4; ++i) p[i] = *(const v4f*)(src + i * 4);
}
__device__ __forceinline__ void cvtstore16(const v4f* p, _Float16* dst) {
  v8h h0, h1;
#pragma unroll
  for (int j = 0; j < 4; ++j) {
    h0[j]     = (_Float16)p[0][j];
    h0[j + 4] = (_Float16)p[1][j];
    h1[j]     = (_Float16)p[2][j];
    h1[j + 4] = (_Float16)p[3][j];
  }
  *(v8h*)(dst)     = h0;
  *(v8h*)(dst + 8) = h1;
}

// ===================================================================
// Kernel 1: mixed = clip(hidden @ Wqkv^T); split into Q,K (B,NH,S,HD)
//           and V pre-transposed (B,NH,HD,S). M=4096, N=6144, K=2048.
// 8 waves arranged 4(M)x2(N); wave tile 32x64 = 2 A-frags x 4 B-frags.
// Global loads software-pipelined one k-step ahead (fp32->f16 convert).
// ===================================================================
__global__ __launch_bounds__(256)
void qkv_gemm(const float* __restrict__ hidden, const float* __restrict__ Wqkv,
              _Float16* __restrict__ qws, _Float16* __restrict__ kws,
              _Float16* __restrict__ vws) {
  __shared__ __align__(16) _Float16 As[128][40];   // [m][k] f16 (80B stride)
  __shared__ __align__(16) _Float16 Bs[128][40];   // [n][k] f16
  const int nb = blockIdx.x, mb = blockIdx.y;
  const int t = threadIdx.x;
  const int w = t >> 5, lane = t & 31, lm = lane & 15, g = lane >> 4;
  const int wm = w >> 1, wn = w & 1;               // 4x2 wave grid
  const int frow = t >> 1, fhalf = t & 1;

  const float* srcA0 = hidden + (size_t)(mb * 128 + frow) * kH + fhalf * 16;
  const float* srcB0 = Wqkv   + (size_t)(nb * 128 + frow) * kH + fhalf * 16;

  v8f acc[2][4];
#pragma unroll
  for (int mt = 0; mt < 2; ++mt)
#pragma unroll
    for (int nt = 0; nt < 4; ++nt)
#pragma unroll
      for (int r = 0; r < 8; ++r) acc[mt][nt][r] = 0.0f;

  v4f pa[4], pb[4];
  gload16(srcA0, pa);
  gload16(srcB0, pb);

  for (int k0 = 0; k0 < kH; k0 += 32) {
    cvtstore16(pa, &As[frow][fhalf * 16]);
    cvtstore16(pb, &Bs[frow][fhalf * 16]);
    __syncthreads();
    if (k0 + 32 < kH) {            // prefetch next k-block (overlaps WMMAs)
      gload16(srcA0 + k0 + 32, pa);
      gload16(srcB0 + k0 + 32, pb);
    }
    AFrag a[2], bf[4];
#pragma unroll
    for (int mt = 0; mt < 2; ++mt) {
      const _Float16* ap = &As[wm * 32 + mt * 16 + lm][0];
      a[mt].h[0] = *(const v8h*)(ap + g * 8);
      a[mt].h[1] = *(const v8h*)(ap + 16 + g * 8);
    }
#pragma unroll
    for (int nt = 0; nt < 4; ++nt) {
      const _Float16* bp = &Bs[wn * 64 + nt * 16 + lm][g * 8];
      bf[nt].h[0] = *(const v8h*)bp;
      bf[nt].h[1] = *(const v8h*)(bp + 16);
    }
#pragma unroll
    for (int mt = 0; mt < 2; ++mt)
#pragma unroll
      for (int nt = 0; nt < 4; ++nt)
        acc[mt][nt] = wmma16(a[mt], bf[nt], acc[mt][nt]);
    __syncthreads();
  }

  // C layout: lane=N(mod16), row M = (lane/16)*8 + r
#pragma unroll
  for (int mt = 0; mt < 2; ++mt) {
    const int m0 = mb * 128 + wm * 32 + mt * 16 + g * 8;
#pragma unroll
    for (int nt = 0; nt < 4; ++nt) {
      const int ng    = nb * 128 + wn * 64 + nt * 16 + lm;
      const int which = ng >> 11;        // 0=Q 1=K 2=V
      const int hc    = ng & (kH - 1);
      const int hh    = hc >> 7, d = hc & (kHD - 1);
#pragma unroll
      for (int r = 0; r < 8; ++r) {
        const int m = m0 + r;
        const int bb = m >> 11, s = m & (kS - 1);
        float vv = fminf(kClip, fmaxf(-kClip, acc[mt][nt][r]));
        const _Float16 hv = (_Float16)vv;
        const size_t bh = (size_t)bb * kNH + hh;
        if (which == 0)      qws[(bh * kS + s) * kHD + d] = hv;
        else if (which == 1) kws[(bh * kS + s) * kHD + d] = hv;
        else                 vws[(bh * kHD + d) * kS + s] = hv;  // transposed
      }
    }
  }
}

// ===================================================================
// Kernel 2: flash attention. One WG = one (b,h) x 128 query rows.
// 8 waves x 16 rows. 64-key tiles DMA'd into double-buffered LDS via
// GLOBAL_LOAD_ASYNC_TO_LDS_B128 one tile ahead (ASYNCcnt pipelining).
// Dynamic LDS: 2x K-tile + 2x V^T-tile + P buffer = 88 KB.
// ===================================================================
constexpr int kKsStride = 136;                 // [key][hd], 272B rows
constexpr int kVtStride = 72;                  // [hd][key], 144B rows
constexpr int kKsTile   = 64 * kKsStride;      // elems per K buffer
constexpr int kVtTile   = 128 * kVtStride;     // elems per V buffer
constexpr int kSmemElems = 2 * kKsTile + 2 * kVtTile + 128 * kVtStride;

__global__ __launch_bounds__(256)
void flash_attn(const _Float16* __restrict__ qws, const _Float16* __restrict__ kws,
                const _Float16* __restrict__ vws, const float* __restrict__ bias,
                _Float16* __restrict__ ctx) {
  extern __shared__ _Float16 smem[];
  _Float16* KsB = smem;                        // [2][64][136]
  _Float16* VtB = smem + 2 * kKsTile;          // [2][128][72]
  _Float16* Ps  = smem + 2 * kKsTile + 2 * kVtTile;  // [128][72]

  const int qb = blockIdx.x, hh = blockIdx.y, bz = blockIdx.z;
  const int t = threadIdx.x;
  const int w = t >> 5, lane = t & 31, lm = lane & 15, g = lane >> 4;
  const size_t bh = (size_t)bz * kNH + hh;
  const int q0 = qb * 128 + w * 16 + g * 8;

  // Q fragments (16 rows x 128 hd) kept in registers: 4 chunks of K=32
  AFrag qf[4];
  {
    const _Float16* qp = qws + (bh * kS + (size_t)qb * 128 + w * 16 + lm) * kHD;
#pragma unroll
    for (int c = 0; c < 4; ++c) {
      qf[c].h[0] = *(const v8h*)(qp + c * 32 + g * 8);
      qf[c].h[1] = *(const v8h*)(qp + c * 32 + 16 + g * 8);
    }
  }

  float mrow[8], lrow[8];
  v8f oacc[8];
#pragma unroll
  for (int r = 0; r < 8; ++r) { mrow[r] = -3.0e38f; lrow[r] = 0.0f; }
#pragma unroll
  for (int dt = 0; dt < 8; ++dt)
#pragma unroll
    for (int r = 0; r < 8; ++r) oacc[dt][r] = 0.0f;

  // async tile staging coordinates (per thread: 4x16B of K, 4x16B of V)
  const int rowK = t >> 2, q4 = t & 3;   // K: 64 rows x 128 hd
  const int rowV = t >> 1, hf = t & 1;   // V^T: 128 rows x 64 keys
  const _Float16* srcKb = kws + (bh * kS + rowK) * kHD + q4 * 32;
  const _Float16* srcVb = vws + (bh * kHD + rowV) * kS + hf * 32;

  auto issue_tile = [&](int kt, int buf) {
    const _Float16* sK = srcKb + (size_t)kt * 64 * kHD;
    const _Float16* sV = srcVb + (size_t)kt * 64;
    _Float16* dK = KsB + buf * kKsTile + rowK * kKsStride + q4 * 32;
    _Float16* dV = VtB + buf * kVtTile + rowV * kVtStride + hf * 32;
#pragma unroll
    for (int i = 0; i < 4; ++i) {
      async_ld_b128(dK + i * 8, sK + i * 8);
      async_ld_b128(dV + i * 8, sV + i * 8);
    }
  };

  const int ktmax = 2 * qb + 1;          // causal: keys <= last row of block
  issue_tile(0, 0);
  wait_async0();
  __syncthreads();

  for (int kt = 0; kt <= ktmax; ++kt) {
    const int cur = kt & 1;
    if (kt < ktmax) {
      issue_tile(kt + 1, cur ^ 1);       // DMA overlaps the WMMA phase below
      // prefetch next tile's position-bias rows (dominant HBM stream)
#pragma unroll
      for (int r = 0; r < 8; ++r)
        __builtin_prefetch(bias + ((size_t)hh * kS + q0 + r) * kS +
                               (size_t)(kt + 1) * 64 + lm * 4, 0, 3);
    }
    const _Float16* Ks = KsB + cur * kKsTile;
    const _Float16* Vt = VtB + cur * kVtTile;

    // scores: 16 rows x 64 keys; batch B-frags per hd-chunk, then WMMAs
    v8f sacc[4];
#pragma unroll
    for (int nt = 0; nt < 4; ++nt)
#pragma unroll
      for (int r = 0; r < 8; ++r) sacc[nt][r] = 0.0f;
#pragma unroll
    for (int c = 0; c < 4; ++c) {
      AFrag bf[4];
#pragma unroll
      for (int nt = 0; nt < 4; ++nt) {
        const _Float16* kp = Ks + (nt * 16 + lm) * kKsStride + c * 32 + g * 8;
        bf[nt].h[0] = *(const v8h*)kp;
        bf[nt].h[1] = *(const v8h*)(kp + 16);
      }
#pragma unroll
      for (int nt = 0; nt < 4; ++nt)
        sacc[nt] = wmma16(qf[c], bf[nt], sacc[nt]);
    }

    // scale + position bias + causal mask
    const bool need_mask = (kt >= 2 * qb);
#pragma unroll
    for (int r = 0; r < 8; ++r) {
      const int q = q0 + r;
      const float* bp = bias + ((size_t)hh * kS + q) * kS + (size_t)kt * 64;
#pragma unroll
      for (int nt = 0; nt < 4; ++nt) {
        const int kc = nt * 16 + lm;
        float sv = sacc[nt][r] * kScale + bp[kc];
        if (need_mask && (kt * 64 + kc) > q) sv = -3.0e38f;
        sacc[nt][r] = sv;
      }
    }

    // online softmax (row stats shared across the 16-lane half-wave)
#pragma unroll
    for (int r = 0; r < 8; ++r) {
      float tm = sacc[0][r];
#pragma unroll
      for (int nt = 1; nt < 4; ++nt) tm = fmaxf(tm, sacc[nt][r]);
      tm = fmaxf(tm, __shfl_xor(tm, 1, 32));
      tm = fmaxf(tm, __shfl_xor(tm, 2, 32));
      tm = fmaxf(tm, __shfl_xor(tm, 4, 32));
      tm = fmaxf(tm, __shfl_xor(tm, 8, 32));
      const float mn = fmaxf(mrow[r], tm);
      const float alpha = __expf(mrow[r] - mn);
      mrow[r] = mn;
      float rs = 0.0f;
#pragma unroll
      for (int nt = 0; nt < 4; ++nt) {
        float p = __expf(sacc[nt][r] - mn);
        sacc[nt][r] = p;
        rs += p;
      }
      rs += __shfl_xor(rs, 1, 32);
      rs += __shfl_xor(rs, 2, 32);
      rs += __shfl_xor(rs, 4, 32);
      rs += __shfl_xor(rs, 8, 32);
      lrow[r] = lrow[r] * alpha + rs;
#pragma unroll
      for (int dt = 0; dt < 8; ++dt) oacc[dt][r] *= alpha;
    }

    // C-layout -> A-layout for P via wave-private LDS region
#pragma unroll
    for (int r = 0; r < 8; ++r)
#pragma unroll
      for (int nt = 0; nt < 4; ++nt)
        Ps[(w * 16 + g * 8 + r) * kVtStride + nt * 16 + lm] = (_Float16)sacc[nt][r];

    // O += P(16x64) · V(64x128): batch 4 V-frags, then 4 WMMAs
#pragma unroll
    for (int c = 0; c < 2; ++c) {
      AFrag pa;
      const _Float16* pp = Ps + (w * 16 + lm) * kVtStride + c * 32;
      pa.h[0] = *(const v8h*)(pp + g * 8);
      pa.h[1] = *(const v8h*)(pp + 16 + g * 8);
#pragma unroll
      for (int h4 = 0; h4 < 2; ++h4) {
        AFrag bf[4];
#pragma unroll
        for (int j = 0; j < 4; ++j) {
          const _Float16* vp = Vt + ((h4 * 4 + j) * 16 + lm) * kVtStride + c * 32 + g * 8;
          bf[j].h[0] = *(const v8h*)vp;
          bf[j].h[1] = *(const v8h*)(vp + 16);
        }
#pragma unroll
        for (int j = 0; j < 4; ++j)
          oacc[h4 * 4 + j] = wmma16(pa, bf[j], oacc[h4 * 4 + j]);
      }
    }

    wait_async0();                 // next tile's DMA done (issued pre-compute)
    __syncthreads();               // everyone finished reading buffer `cur`
  }

  // normalize and store ctx as [B,S,H] f16 (heads re-merged on store)
#pragma unroll
  for (int r = 0; r < 8; ++r) {
    const float inv = 1.0f / lrow[r];
    const int q = q0 + r;
    _Float16* cp = ctx + ((size_t)bz * kS + q) * kH + hh * kHD;
#pragma unroll
    for (int dt = 0; dt < 8; ++dt)
      cp[dt * 16 + lm] = (_Float16)(oacc[dt][r] * inv);
  }
}

// ===================================================================
// Kernel 3: out = ctx @ Wout^T.  M=4096, N=2048, K=2048, fp32 output.
// Same 4(M)x2(N) wave tiling and global->LDS pipelining.
// ===================================================================
__global__ __launch_bounds__(256)
void out_gemm(const _Float16* __restrict__ ctx, const float* __restrict__ Wout,
              float* __restrict__ out) {
  __shared__ __align__(16) _Float16 As[128][40];
  __shared__ __align__(16) _Float16 Bs[128][40];
  const int nb = blockIdx.x, mb = blockIdx.y;
  const int t = threadIdx.x;
  const int w = t >> 5, lane = t & 31, lm = lane & 15, g = lane >> 4;
  const int wm = w >> 1, wn = w & 1;
  const int frow = t >> 1, fhalf = t & 1;

  const _Float16* srcA0 = ctx + (size_t)(mb * 128 + frow) * kH + fhalf * 16;
  const float*    srcB0 = Wout + (size_t)(nb * 128 + frow) * kH + fhalf * 16;

  v8f acc[2][4];
#pragma unroll
  for (int mt = 0; mt < 2; ++mt)
#pragma unroll
    for (int nt = 0; nt < 4; ++nt)
#pragma unroll
      for (int r = 0; r < 8; ++r) acc[mt][nt][r] = 0.0f;

  v8h ra[2];
  v4f pb[4];
  ra[0] = *(const v8h*)(srcA0);
  ra[1] = *(const v8h*)(srcA0 + 8);
  gload16(srcB0, pb);

  for (int k0 = 0; k0 < kH; k0 += 32) {
    *(v8h*)&As[frow][fhalf * 16]     = ra[0];
    *(v8h*)&As[frow][fhalf * 16 + 8] = ra[1];
    cvtstore16(pb, &Bs[frow][fhalf * 16]);
    __syncthreads();
    if (k0 + 32 < kH) {
      ra[0] = *(const v8h*)(srcA0 + k0 + 32);
      ra[1] = *(const v8h*)(srcA0 + k0 + 40);
      gload16(srcB0 + k0 + 32, pb);
    }
    AFrag a[2], bf[4];
#pragma unroll
    for (int mt = 0; mt < 2; ++mt) {
      const _Float16* ap = &As[wm * 32 + mt * 16 + lm][0];
      a[mt].h[0] = *(const v8h*)(ap + g * 8);
      a[mt].h[1] = *(const v8h*)(ap + 16 + g * 8);
    }
#pragma unroll
    for (int nt = 0; nt < 4; ++nt) {
      const _Float16* bp = &Bs[wn * 64 + nt * 16 + lm][g * 8];
      bf[nt].h[0] = *(const v8h*)bp;
      bf[nt].h[1] = *(const v8h*)(bp + 16);
    }
#pragma unroll
    for (int mt = 0; mt < 2; ++mt)
#pragma unroll
      for (int nt = 0; nt < 4; ++nt)
        acc[mt][nt] = wmma16(a[mt], bf[nt], acc[mt][nt]);
    __syncthreads();
  }

#pragma unroll
  for (int mt = 0; mt < 2; ++mt) {
    const int m0 = mb * 128 + wm * 32 + mt * 16 + g * 8;
#pragma unroll
    for (int nt = 0; nt < 4; ++nt) {
      const int ng = nb * 128 + wn * 64 + nt * 16 + lm;
#pragma unroll
      for (int r = 0; r < 8; ++r)
        out[(size_t)(m0 + r) * kH + ng] = acc[mt][nt][r];
    }
  }
}

// ===================================================================
extern "C" void kernel_launch(void* const* d_in, const int* in_sizes, int n_in,
                              void* d_out, int out_size, void* d_ws, size_t ws_size,
                              hipStream_t stream) {
  (void)in_sizes; (void)n_in; (void)out_size; (void)ws_size;
  const float* hidden = (const float*)d_in[0];
  const float* bias   = (const float*)d_in[1];
  // d_in[2] is the boolean causal mask; it is exactly the upper triangle,
  // so it is recomputed analytically inside flash_attn.
  const float* Wqkv   = (const float*)d_in[3];
  const float* Wout   = (const float*)d_in[4];
  float* out = (float*)d_out;

  const size_t E = (size_t)kB * kNH * kS * kHD;   // 8,388,608 elems per tensor
  _Float16* qws = (_Float16*)d_ws;                // [B,NH,S,HD]
  _Float16* kws = qws + E;                        // [B,NH,S,HD]
  _Float16* vws = kws + E;                        // [B,NH,HD,S] (transposed)
  _Float16* ctx = vws + E;                        // [B,S,H]

  const int smem_bytes = kSmemElems * (int)sizeof(_Float16);   // ~88 KB
  (void)hipFuncSetAttribute((const void*)flash_attn,
                            hipFuncAttributeMaxDynamicSharedMemorySize,
                            smem_bytes);

  qkv_gemm <<<dim3(3 * kH / 128, (kB * kS) / 128), 256, 0, stream>>>(hidden, Wqkv, qws, kws, vws);
  flash_attn<<<dim3(kS / 128, kNH, kB), 256, smem_bytes, stream>>>(qws, kws, vws, bias, ctx);
  out_gemm <<<dim3(kH / 128, (kB * kS) / 128),     256, 0, stream>>>(ctx, Wout, out);
}
